// SelfAttention_3831110828068
// MI455X (gfx1250) — matrix operand order repaired and checked by reading, not compiled, not run on hardware
//
#include <hip/hip_runtime.h>

// ---------------------------------------------------------------------------
// CDNA5 (gfx1250) fused self-attention with pooled KV.
// All GEMMs via v_wmma_f32_16x16x32_bf16 (wave32). f32 softmax.
// K panel staged into LDS with global_load_async_to_lds_b128 (ASYNCcnt).
// ---------------------------------------------------------------------------

#define DI __device__ __forceinline__

typedef __attribute__((ext_vector_type(16))) __bf16 v16bf;
typedef __attribute__((ext_vector_type(8)))  float  v8f;

union FragU { v16bf v; unsigned u[8]; unsigned short s[16]; };

constexpr int Bn = 16;    // batch
constexpr int Cc = 256;   // channels
constexpr int C8 = 32;    // qk dim
constexpr int Nn = 4096;  // H*W
constexpr int Mm = 1024;  // pooled positions
constexpr int KV = 288;   // 32 (K) + 256 (V) conv rows
constexpr int ROWS = 320; // 32 Q + 32 K + 256 V output rows
constexpr int QT = 32;    // query tile of the attention kernel
constexpr int PLD = 1040; // padded LDS stride (bank-conflict avoidance)

DI unsigned short f2bf(float f) {                // round-to-nearest-even
  unsigned u = __builtin_bit_cast(unsigned, f);
  u += 0x7FFFu + ((u >> 16) & 1u);
  return (unsigned short)(u >> 16);
}
DI float bf2f(unsigned short h) {
  unsigned u = ((unsigned)h) << 16;
  return __builtin_bit_cast(float, u);
}

// A-matrix 16x32 bf16 K-interleave per ISA 7.12.2:
// VGPR p holds K pair {a_koff(p,hi), +1}; hi = lane/16 adds +8 within blocks.
DI int a_koff(int p, int hi) { return ((p < 4) ? 2 * p : 2 * p + 8) + 8 * hi; }

// A fragment from row-major [row][k] storage (ld in ushorts, k pairs even).
DI v16bf load_a(const unsigned short* base, int ld, int row0, int k0, int lane) {
  FragU f;
  const int hi = lane >> 4;
  const unsigned short* r = base + (long)(row0 + (lane & 15)) * ld + k0;
#pragma unroll
  for (int p = 0; p < 8; ++p)
    f.u[p] = *(const unsigned*)(r + a_koff(p, hi));
  return f.v;
}

// B fragment: B(k, col) taken from [col][k] storage; lane -> col = lane%16,
// K = a_koff(p, hi) (same interleave as A), K pairs contiguous in memory.
DI v16bf load_b(const unsigned short* base, int ld, int col0, int k0, int lane) {
  FragU f;
  const int hi = lane >> 4;
  const unsigned short* c = base + (long)(col0 + (lane & 15)) * ld + k0;
#pragma unroll
  for (int p = 0; p < 8; ++p)
    f.u[p] = *(const unsigned*)(c + a_koff(p, hi));
  return f.v;
}

DI v8f wmma_bf16(v16bf a, v16bf b, v8f c) {
  return __builtin_amdgcn_wmma_f32_16x16x32_bf16(false, a, false, b, (short)0, c,
                                                 false, false);
}

// Async copy 16B global -> LDS (tracked by ASYNCcnt).
DI void async_copy_b128(const void* gptr, void* lptr) {
  unsigned lds_off = (unsigned)(unsigned long long)lptr;  // LDS aperture: low 32b
  unsigned long long ga = (unsigned long long)gptr;
  asm volatile("global_load_async_to_lds_b128 %0, %1, off"
               :: "v"(lds_off), "v"(ga) : "memory");
}
DI void wait_async0() { asm volatile("s_wait_asynccnt 0" ::: "memory"); }

// ---------------------------------------------------------------------------
// K0: concat Wq|Wk|Wv and convert to bf16 (320 x 256)
// ---------------------------------------------------------------------------
__global__ void wcat_kernel(const float* __restrict__ Wq, const float* __restrict__ Wk,
                            const float* __restrict__ Wv, unsigned short* __restrict__ wcat) {
  int idx = blockIdx.x * blockDim.x + threadIdx.x;
  if (idx >= ROWS * Cc) return;
  int row = idx / Cc, c = idx % Cc;
  float v = (row < 32) ? Wq[row * Cc + c]
          : (row < 64) ? Wk[(row - 32) * Cc + c]
                       : Wv[(row - 64) * Cc + c];
  wcat[idx] = f2bf(v);
}

// ---------------------------------------------------------------------------
// K1: fused QKV 1x1-conv GEMM: Wcat(320x256) x x_b(256x64-tile) via WMMA.
// Q -> qb[b][n][32] bf16 (A-layout for attention), K/V conv -> cb[b][288][n].
// ---------------------------------------------------------------------------
__global__ __launch_bounds__(256) void proj_kernel(
    const float* __restrict__ x, const unsigned short* __restrict__ wcat,
    const float* __restrict__ bq, const float* __restrict__ bk,
    const float* __restrict__ bv, unsigned short* __restrict__ qb,
    unsigned short* __restrict__ cb) {
  __shared__ __align__(16) unsigned short xT[64][264];  // x tile transposed, bf16

  const int b = blockIdx.y;
  const int n0 = blockIdx.x * 64;
  const int tid = threadIdx.x, lane = tid & 31, wave = tid >> 5;

  // Stage x[b, :, n0:n0+64] into LDS transposed ([col][c]) as bf16.
  const float* xb = x + (long)b * Cc * Nn + n0;
  for (int idx = tid; idx < Cc * 64; idx += 256) {
    int c = idx >> 6, j = idx & 63;
    xT[j][c] = f2bf(xb[(long)c * Nn + j]);
  }
  __syncthreads();

  const int hi = lane >> 4, lo = lane & 15;
  // 20 row-tiles x 4 col-tiles = 80 WMMA output tiles; 8 waves.
  for (int t = wave; t < 80; t += 8) {
    const int row0 = (t % 20) * 16;
    const int col0 = (t / 20) * 16;
    v8f acc = {0.f, 0.f, 0.f, 0.f, 0.f, 0.f, 0.f, 0.f};
#pragma unroll
    for (int ks = 0; ks < 8; ++ks) {
      v16bf a = load_a(wcat, Cc, row0, ks * 32, lane);
      v16bf bm = load_b(&xT[0][0], 264, col0, ks * 32, lane);
      acc = wmma_bf16(a, bm, acc);
    }
    const int n = n0 + col0 + lo;
    if (row0 < 32) {  // Q rows: pack 8 contiguous o-channels, one b128 store
      unsigned short tmp[8];
#pragma unroll
      for (int r = 0; r < 8; ++r) {
        int orow = row0 + 8 * hi + r;
        tmp[r] = f2bf(acc[r] + bq[orow]);
      }
      *(uint4*)&qb[((long)(b * Nn + n)) * C8 + row0 + 8 * hi] = *(const uint4*)tmp;
    } else {          // K/V conv rows (pooled later)
#pragma unroll
      for (int r = 0; r < 8; ++r) {
        int orow = row0 + 8 * hi + r;
        float bias = (orow < 64) ? bk[orow - 32] : bv[orow - 64];
        cb[((long)(b * KV + (orow - 32))) * Nn + n] = f2bf(acc[r] + bias);
      }
    }
  }
}

// ---------------------------------------------------------------------------
// K2: 2x2 max pool on conv(K)/conv(V); emit B-fragment-friendly layouts:
//     kb[b][m][32] (scores B), vb[b][c][m] (PV B).
// ---------------------------------------------------------------------------
__global__ void pool_kernel(const unsigned short* __restrict__ cb,
                            unsigned short* __restrict__ kb,
                            unsigned short* __restrict__ vb) {
  long idx = (long)blockIdx.x * blockDim.x + threadIdx.x;
  if (idx >= (long)Bn * KV * Mm) return;
  int m = (int)(idx % Mm);
  int ch = (int)((idx / Mm) % KV);
  int b = (int)(idx / ((long)Mm * KV));
  int i = m >> 5, j = m & 31;                      // pooled (i,j) of 32x32
  const unsigned short* s = cb + ((long)(b * KV + ch)) * Nn + i * 128 + j * 2;
  float mx = fmaxf(fmaxf(bf2f(s[0]), bf2f(s[1])),
                   fmaxf(bf2f(s[64]), bf2f(s[65])));
  unsigned short o = f2bf(mx);
  if (ch < 32) kb[((long)(b * Mm + m)) * C8 + ch] = o;
  else         vb[((long)(b * Cc + (ch - 32))) * Mm + m] = o;
}

// ---------------------------------------------------------------------------
// K3: fused attention per 32-query tile:
//   K panel (64 KB) async-copied to LDS (global_load_async_to_lds_b128)
//   S = Q·K (one WMMA per 16x16 tile, K-dim = 32) -> LDS f32
//   row softmax (wave32 shfl reductions) -> P bf16 in LDS
//   out = P·Vᵀ (32 WMMA k-steps per 16x16 tile), epilogue gamma*out + x.
// ---------------------------------------------------------------------------
__global__ __launch_bounds__(256) void attn_kernel(
    const unsigned short* __restrict__ qb, const unsigned short* __restrict__ kb,
    const unsigned short* __restrict__ vb, const float* __restrict__ x,
    const float* __restrict__ gamma, float* __restrict__ out) {
  __shared__ __align__(16) unsigned short Klds[Mm * C8];  //  64 KB K panel
  __shared__ __align__(16) float Ssm[QT][PLD];            // 130 KB scores
  __shared__ __align__(16) unsigned short Psm[QT][PLD];   //  65 KB probs (bf16)

  const int b = blockIdx.y;
  const int n0 = blockIdx.x * QT;
  const int tid = threadIdx.x, lane = tid & 31, wave = tid >> 5;
  const int hi = lane >> 4, lo = lane & 15;

  // ---- async-stage K[b] (1024 x 32 bf16) into LDS: 4096 x b128 chunks ----
  const unsigned short* kbb = kb + (long)b * Mm * C8;
#pragma unroll
  for (int it = 0; it < 16; ++it) {
    int i = tid + it * 256;                       // chunk index
    async_copy_b128(kbb + i * 8, &Klds[i * 8]);
  }

  // ---- Q fragments for both 16-row query tiles (overlaps async copy) ----
  const v16bf qa0 = load_a(qb + (long)(b * Nn + n0) * C8, C8, 0, 0, lane);
  const v16bf qa1 = load_a(qb + (long)(b * Nn + n0 + 16) * C8, C8, 0, 0, lane);

  wait_async0();
  __syncthreads();

  // ---- stage 1: S[q][m] ----
  for (int kt = wave; kt < 64; kt += 8) {
    const int m0 = kt * 16;
    v16bf kf = load_b(Klds, C8, m0, 0, lane);
    v8f z = {0.f, 0.f, 0.f, 0.f, 0.f, 0.f, 0.f, 0.f};
    v8f s0 = wmma_bf16(qa0, kf, z);
    v8f s1 = wmma_bf16(qa1, kf, z);
#pragma unroll
    for (int r = 0; r < 8; ++r) {
      Ssm[8 * hi + r][m0 + lo] = s0[r];
      Ssm[16 + 8 * hi + r][m0 + lo] = s1[r];
    }
  }
  __syncthreads();

  // ---- stage 2: softmax over M (4 rows per wave) ----
#pragma unroll
  for (int rr = 0; rr < 4; ++rr) {
    const int row = wave * 4 + rr;
    float mx = -3.402823466e38f;
    for (int m = lane; m < Mm; m += 32) mx = fmaxf(mx, Ssm[row][m]);
#pragma unroll
    for (int off = 16; off; off >>= 1) mx = fmaxf(mx, __shfl_xor(mx, off, 32));
    float sum = 0.f;
    for (int m = lane; m < Mm; m += 32) {
      float e = __expf(Ssm[row][m] - mx);
      Ssm[row][m] = e;
      sum += e;
    }
#pragma unroll
    for (int off = 16; off; off >>= 1) sum += __shfl_xor(sum, off, 32);
    const float inv = 1.f / sum;
    for (int m = lane; m < Mm; m += 32) Psm[row][m] = f2bf(Ssm[row][m] * inv);
  }
  __syncthreads();

  // ---- stage 3: out = P (32 x M) x Vᵀ (M x 256): 2 q-tiles x 16 c-tiles ----
  const float g = gamma[0];
  const unsigned short* vbb = vb + (long)b * Cc * Mm;
  for (int t = wave; t < 32; t += 8) {
    const int qt = t & 1;
    const int c0 = (t >> 1) * 16;
    v8f acc = {0.f, 0.f, 0.f, 0.f, 0.f, 0.f, 0.f, 0.f};
    for (int ks = 0; ks < 32; ++ks) {
      if (ks + 1 < 32)  // global_prefetch_b8 on the V stream
        __builtin_prefetch(vbb + (long)(c0 + lo) * Mm + (ks + 1) * 32 + 16 * hi, 0, 0);
      v16bf pa = load_a(&Psm[qt * 16][0], PLD, 0, ks * 32, lane);
      v16bf vf = load_b(vbb, Mm, c0, ks * 32, lane);
      acc = wmma_bf16(pa, vf, acc);
    }
    // D layout: per lane 8 consecutive n positions, fixed channel -> float4 I/O
    const int ch = c0 + lo;
    const long base = ((long)(b * Cc + ch)) * Nn + n0 + qt * 16 + 8 * hi;
    float4 x0 = *(const float4*)&x[base];
    float4 x1 = *(const float4*)&x[base + 4];
    float4 o0, o1;
    o0.x = g * acc[0] + x0.x; o0.y = g * acc[1] + x0.y;
    o0.z = g * acc[2] + x0.z; o0.w = g * acc[3] + x0.w;
    o1.x = g * acc[4] + x1.x; o1.y = g * acc[5] + x1.y;
    o1.z = g * acc[6] + x1.z; o1.w = g * acc[7] + x1.w;
    *(float4*)&out[base] = o0;
    *(float4*)&out[base + 4] = o1;
  }
}

// ---------------------------------------------------------------------------
extern "C" void kernel_launch(void* const* d_in, const int* in_sizes, int n_in,
                              void* d_out, int out_size, void* d_ws, size_t ws_size,
                              hipStream_t stream) {
  const float* x     = (const float*)d_in[0];
  const float* Wq    = (const float*)d_in[1];
  const float* bq    = (const float*)d_in[2];
  const float* Wk    = (const float*)d_in[3];
  const float* bk    = (const float*)d_in[4];
  const float* Wv    = (const float*)d_in[5];
  const float* bv    = (const float*)d_in[6];
  const float* gamma = (const float*)d_in[7];
  float* out = (float*)d_out;

  // Workspace carve-up (bf16 ushorts): wcat | qb | cb | kb | vb  (~49 MB)
  unsigned short* wcat = (unsigned short*)d_ws;
  unsigned short* qb = wcat + (size_t)ROWS * Cc;
  unsigned short* cb = qb + (size_t)Bn * Nn * C8;
  unsigned short* kb = cb + (size_t)Bn * KV * Nn;
  unsigned short* vb = kb + (size_t)Bn * Mm * C8;

  wcat_kernel<<<(ROWS * Cc + 255) / 256, 256, 0, stream>>>(Wq, Wk, Wv, wcat);
  proj_kernel<<<dim3(Nn / 64, Bn), 256, 0, stream>>>(x, wcat, bq, bk, bv, qb, cb);
  pool_kernel<<<(int)(((long)Bn * KV * Mm + 255) / 256), 256, 0, stream>>>(cb, kb, vb);
  attn_kernel<<<dim3(Nn / QT, Bn), 256, 0, stream>>>(qb, kb, vb, x, gamma, out);
}
